// FullQKAttention_59167469470174
// MI455X (gfx1250) — matrix operand order repaired and checked by reading, not compiled, not run on hardware
//
#include <hip/hip_runtime.h>
#include <hip/hip_bf16.h>

// FullQKAttention for MI455X (gfx1250): flash-attention with f16 WMMA.
// b=8, n=4096, d=128.  out = softmax(mask_diag(Q K^T / sqrt(d))) V,
// Q = qk, K = qk / ||qk||_2, V = v.
//
// Workspace layout (needs 24 MB):
//   qf : f16 [8][4096][128]  (Q * d^-0.5)
//   kf : f16 [8][4096][128]  (normalized K)
//   vt : f16 [8][128][4096]  (V transposed per batch)

typedef __attribute__((ext_vector_type(16))) _Float16 v16h;
typedef __attribute__((ext_vector_type(4)))  _Float16 v4h;
typedef __attribute__((ext_vector_type(8)))  float    v8f;
typedef __attribute__((ext_vector_type(4)))  unsigned int u32x4;
typedef int v4i_vs __attribute__((vector_size(16)));

union FragU { u32x4 q[2]; v16h h; };

#define BATCH 8
#define SEQ   4096
#define DIM   128
#define QK_SCALE 0.08838834764831845f   // 1/sqrt(128)
#define NEG_BIG  (-3.0e38f)

// Async global->LDS staging (CDNA5 path, ASYNCcnt-tracked).
// Probe-verified signature: (v4i AS1* gsrc, v4i AS3* ldst, imm offset, imm cpol)
#if defined(__gfx1250__) && __has_builtin(__builtin_amdgcn_global_load_async_to_lds_b128)
#define HAVE_ASYNC_LDS 1
#define ASYNC_COPY_B128(g, l)                                              \
    __builtin_amdgcn_global_load_async_to_lds_b128(                        \
        (__attribute__((address_space(1))) v4i_vs*)(g),                    \
        (__attribute__((address_space(3))) v4i_vs*)(l), 0, 0)
#define ASYNC_WAIT() asm volatile("s_wait_asynccnt 0x0" ::: "memory")
#else
#define HAVE_ASYNC_LDS 0
#define ASYNC_WAIT()
#endif

#if defined(__gfx1250__) && __has_builtin(__builtin_amdgcn_sched_group_barrier)
#define SGB(mask, n) __builtin_amdgcn_sched_group_barrier((mask), (n), 0)
#else
#define SGB(mask, n)
#endif

// ---------------------------------------------------------------------------
// Prep: per-row L2 norm, f16 conversion, V transpose.  One wave per row.
// ---------------------------------------------------------------------------
__global__ __launch_bounds__(256) void FullQKAttention_prep(
    const float* __restrict__ qk, const float* __restrict__ v,
    _Float16* __restrict__ qf, _Float16* __restrict__ kf,
    _Float16* __restrict__ vt)
{
    const int lane = threadIdx.x & 31;
    const int row  = blockIdx.x * 8 + (threadIdx.x >> 5);   // 0 .. B*N-1
    const size_t base = (size_t)row * DIM;

    const float4 q = ((const float4*)(qk + base))[lane];
    float ss = q.x*q.x + q.y*q.y + q.z*q.z + q.w*q.w;
    ss += __shfl_xor(ss, 16);
    ss += __shfl_xor(ss, 8);
    ss += __shfl_xor(ss, 4);
    ss += __shfl_xor(ss, 2);
    ss += __shfl_xor(ss, 1);
    const float inv = 1.0f / fmaxf(sqrtf(ss), 1e-12f);

    v4h qh = { (_Float16)(q.x * QK_SCALE), (_Float16)(q.y * QK_SCALE),
               (_Float16)(q.z * QK_SCALE), (_Float16)(q.w * QK_SCALE) };
    v4h kh = { (_Float16)(q.x * inv), (_Float16)(q.y * inv),
               (_Float16)(q.z * inv), (_Float16)(q.w * inv) };
    *(v4h*)(qf + base + lane * 4) = qh;
    *(v4h*)(kf + base + lane * 4) = kh;

    const float4 vv = ((const float4*)(v + base))[lane];
    const int b = row >> 12;
    const int n = row & (SEQ - 1);
    const size_t vbase = ((size_t)b * DIM + lane * 4) * SEQ + n;
    vt[vbase          ] = (_Float16)vv.x;
    vt[vbase +     SEQ] = (_Float16)vv.y;
    vt[vbase + 2 * SEQ] = (_Float16)vv.z;
    vt[vbase + 3 * SEQ] = (_Float16)vv.w;
}

// ---------------------------------------------------------------------------
// Flash attention: 8 waves / block, each wave owns 16 query rows (128/block).
// Key loop in tiles of 32, double-buffered async LDS staging.
// ---------------------------------------------------------------------------
__global__ __launch_bounds__(256) void FullQKAttention_attn(
    const _Float16* __restrict__ qf, const _Float16* __restrict__ kf,
    const _Float16* __restrict__ vt, float* __restrict__ out)
{
    __shared__ __align__(16) _Float16 Kt0[32 * DIM];   // [key 0..31][d 0..127]
    __shared__ __align__(16) _Float16 Kt1[32 * DIM];
    __shared__ __align__(16) _Float16 Vt0[DIM * 32];   // [d 0..127][key 0..31]
    __shared__ __align__(16) _Float16 Vt1[DIM * 32];
    __shared__ __align__(16) _Float16 Pb [8 * 512];    // per-wave P scratch

    const int tid  = threadIdx.x;
    const int w    = tid >> 5;
    const int lane = tid & 31;
    const int h    = lane >> 4;     // half-wave (0/1)
    const int ln   = lane & 15;
    const int bb   = blockIdx.x >> 5;
    const int m0   = (blockIdx.x & 31) * 128;
    const int rowbase = m0 + w * 16;

    const _Float16* kbat = kf + (size_t)bb * SEQ * DIM;
    const _Float16* vbat = vt + (size_t)bb * DIM * SEQ;
    const int vd   = tid >> 1;          // d-row this thread stages
    const int vhal = tid & 1;           // which 16-key half of that row

    // async-stage one 32-key K tile + V^T tile into the given LDS buffers
    auto stage = [&](int j, _Float16* Kd, _Float16* Vd) {
        const u32x4* ks = (const u32x4*)(kbat + (size_t)j * DIM);
        u32x4*       kd = (u32x4*)Kd;                       // 512 u32x4
        const u32x4* vs = (const u32x4*)(vbat + (size_t)vd * SEQ + j + vhal * 16);
        u32x4*       vw = (u32x4*)(Vd + vd * 32 + vhal * 16);
#if HAVE_ASYNC_LDS
        #pragma unroll
        for (int i = 0; i < 2; ++i)
            ASYNC_COPY_B128(ks + tid + i * 256, kd + tid + i * 256);
        #pragma unroll
        for (int i = 0; i < 2; ++i)
            ASYNC_COPY_B128(vs + i, vw + i);
#else
        #pragma unroll
        for (int i = 0; i < 2; ++i) kd[tid + i * 256] = ks[tid + i * 256];
        #pragma unroll
        for (int i = 0; i < 2; ++i) vw[i] = vs[i];
#endif
    };

    // --- Q A-fragments (16 rows x 128, four 16x32 fragments) ---------------
    FragU qa[4];
    {
        const _Float16* qrow = qf + ((size_t)bb * SEQ + rowbase + ln) * DIM;
        #pragma unroll
        for (int f = 0; f < 4; ++f) {
            qa[f].q[0] = *(const u32x4*)(qrow + f * 32 + h * 16);
            qa[f].q[1] = *(const u32x4*)(qrow + f * 32 + h * 16 + 8);
        }
    }

    v8f acc[8];
    #pragma unroll
    for (int t = 0; t < 8; ++t) acc[t] = {};
    float m[8], l[8];
    #pragma unroll
    for (int r = 0; r < 8; ++r) { m[r] = NEG_BIG; l[r] = 0.0f; }

    _Float16* Pw = Pb + w * 512;

    // one iteration of the key loop: consume (Kc,Vc), stage tile jn into (Kn,Vn)
    auto body = [&](int j0, int jn,
                    const _Float16* Kc, const _Float16* Vc,
                    _Float16* Kn, _Float16* Vn) {
        ASYNC_WAIT();            // our async copies for (Kc,Vc) are done
        __syncthreads();         // everyone's copies visible; prev readers done
        stage(jn, Kn, Vn);       // overlap next-tile DMA with compute below
        __builtin_prefetch(kbat + (size_t)jn * DIM, 0, 0);

        // --- S = Q x K^T : 16 B-fragment ds_loads feeding 2 WMMA chains ---
        FragU kb0[4], kb1[4];
        #pragma unroll
        for (int f = 0; f < 4; ++f) {
            const _Float16* kp0 = Kc + (size_t)ln * DIM + f * 32 + h * 16;
            kb0[f].q[0] = *(const u32x4*)(kp0);
            kb0[f].q[1] = *(const u32x4*)(kp0 + 8);
            const _Float16* kp1 = Kc + (size_t)(16 + ln) * DIM + f * 32 + h * 16;
            kb1[f].q[0] = *(const u32x4*)(kp1);
            kb1[f].q[1] = *(const u32x4*)(kp1 + 8);
        }
        v8f s0 = {}, s1 = {};
        #pragma unroll
        for (int f = 0; f < 4; ++f) {
            s0 = __builtin_amdgcn_wmma_f32_16x16x32_f16(
                     false, qa[f].h, false, kb0[f].h, (short)0, s0, false, false);
            s1 = __builtin_amdgcn_wmma_f32_16x16x32_f16(
                     false, qa[f].h, false, kb1[f].h, (short)0, s1, false, false);
        }

        // --- mask diagonal ------------------------------------------------
        const int col0 = j0 + ln, col1 = j0 + 16 + ln;
        #pragma unroll
        for (int r = 0; r < 8; ++r) {
            const int rowg = rowbase + r + 8 * h;
            if (col0 == rowg) s0[r] = NEG_BIG;
            if (col1 == rowg) s1[r] = NEG_BIG;
        }

        // --- online softmax (row = r + 8h, N striped over 16 lanes) -------
        float al[8];
        #pragma unroll
        for (int r = 0; r < 8; ++r) {
            float mx = fmaxf(s0[r], s1[r]);
            mx = fmaxf(mx, __shfl_xor(mx, 1));
            mx = fmaxf(mx, __shfl_xor(mx, 2));
            mx = fmaxf(mx, __shfl_xor(mx, 4));
            mx = fmaxf(mx, __shfl_xor(mx, 8));
            const float mn = fmaxf(m[r], mx);
            const float a  = __expf(m[r] - mn);
            const float p0 = __expf(s0[r] - mn);
            const float p1 = __expf(s1[r] - mn);
            float rs = p0 + p1;
            rs += __shfl_xor(rs, 1);
            rs += __shfl_xor(rs, 2);
            rs += __shfl_xor(rs, 4);
            rs += __shfl_xor(rs, 8);
            l[r] = l[r] * a + rs;
            m[r] = mn;
            al[r] = a;
            Pw[(r + 8 * h) * 32 +      ln] = (_Float16)p0;
            Pw[(r + 8 * h) * 32 + 16 + ln] = (_Float16)p1;
        }

        // --- rescale accumulators -----------------------------------------
        #pragma unroll
        for (int t = 0; t < 8; ++t) {
            #pragma unroll
            for (int r = 0; r < 8; ++r) acc[t][r] *= al[r];
        }

        // --- reload P as 16x32 A-fragment (per-wave region, no barrier) ---
        FragU pa;
        pa.q[0] = *(const u32x4*)(Pw + ln * 32 + h * 16);
        pa.q[1] = *(const u32x4*)(Pw + ln * 32 + h * 16 + 8);

        // --- acc += P x V -------------------------------------------------
        FragU vb[8];
        #pragma unroll
        for (int t = 0; t < 8; ++t) {
            const _Float16* vp = Vc + (size_t)(t * 16 + ln) * 32 + h * 16;
            vb[t].q[0] = *(const u32x4*)(vp);
            vb[t].q[1] = *(const u32x4*)(vp + 8);
        }
        #pragma unroll
        for (int t = 0; t < 8; ++t)
            acc[t] = __builtin_amdgcn_wmma_f32_16x16x32_f16(
                         false, pa.h, false, vb[t].h, (short)0, acc[t],
                         false, false);

        // --- software pipeline: DS-reads run ahead of each WMMA -----------
        SGB(0x100, 6);                       // 3 K fragments in flight
        #pragma unroll
        for (int i = 0; i < 5; ++i) { SGB(0x008, 1); SGB(0x100, 2); }
        SGB(0x008, 3);                       // finish S WMMAs
        SGB(0x200, 16);                      // P f16 stores
        SGB(0x100, 10);                      // pa + 8 V fragments
        SGB(0x008, 8);                       // PV WMMAs
    };

    stage(0, Kt0, Vt0);
    #pragma unroll 1
    for (int j0 = 0; j0 < SEQ; j0 += 64) {
        const int jn1 = j0 + 32;
        const int jn2 = (j0 + 64 < SEQ) ? j0 + 64 : SEQ - 32;  // clamp (redundant last copy)
        body(j0,  jn1, Kt0, Vt0, Kt1, Vt1);
        body(jn1, jn2, Kt1, Vt1, Kt0, Vt0);
    }

    // --- epilogue: normalize by row sum, store f32 -------------------------
    #pragma unroll
    for (int r = 0; r < 8; ++r) {
        const float inv = 1.0f / l[r];
        const size_t obase =
            ((size_t)bb * SEQ + rowbase + r + 8 * h) * DIM + ln;
        #pragma unroll
        for (int t = 0; t < 8; ++t)
            out[obase + t * 16] = acc[t][r] * inv;
    }
}

// ---------------------------------------------------------------------------
extern "C" void kernel_launch(void* const* d_in, const int* in_sizes, int n_in,
                              void* d_out, int out_size, void* d_ws, size_t ws_size,
                              hipStream_t stream)
{
    (void)in_sizes; (void)n_in; (void)out_size; (void)ws_size;
    const float* qk = (const float*)d_in[0];
    const float* v  = (const float*)d_in[1];
    float* out = (float*)d_out;

    const size_t ELEMS = (size_t)BATCH * SEQ * DIM;     // 4,194,304
    _Float16* qf = (_Float16*)d_ws;
    _Float16* kf = qf + ELEMS;
    _Float16* vt = kf + ELEMS;

    FullQKAttention_prep<<<(BATCH * SEQ) / 8, 256, 0, stream>>>(qk, v, qf, kf, vt);
    FullQKAttention_attn<<<BATCH * (SEQ / 128), 256, 0, stream>>>(qf, kf, vt, out);
}
